// Block_85212151153420
// MI455X (gfx1250) — matrix operand order repaired and checked by reading, not compiled, not run on hardware
//
#include <hip/hip_runtime.h>

typedef __attribute__((ext_vector_type(16))) _Float16 v16h;
typedef __attribute__((ext_vector_type(8)))  float    v8f;

#define N_EMBD  1024
#define N_HEAD  16
#define HEAD    64
#define TSEQ    2048
#define BATCH   2
#define HIDDEN  2048
#define ROWS    (BATCH*TSEQ)   /* 4096 */

union FragU { v16h h; uint4 u[2]; };

__device__ __forceinline__ v8f wmma_f16(v16h a, v16h b, v8f c) {
  // D = A(16x32 f16) * B(32x16 f16) + C(16x16 f32)
  return __builtin_amdgcn_wmma_f32_16x16x32_f16(false, a, false, b, (short)0, c,
                                                false, false);
}

// ---------------------------------------------------------------------------
// LayerNorm: one block (256 thr) per row of 1024; writes f16 normalized row.
// ---------------------------------------------------------------------------
__global__ void ln_kernel(const float* __restrict__ x, const float* __restrict__ w,
                          const float* __restrict__ b, _Float16* __restrict__ out)
{
  const int row = blockIdx.x;
  const int tid = threadIdx.x;
  const float* xr = x + (size_t)row * N_EMBD;
  float s = 0.f, s2 = 0.f;
  for (int i = tid; i < N_EMBD; i += 256) { float v = xr[i]; s += v; s2 += v * v; }
  __shared__ float rs[256], rs2[256];
  rs[tid] = s; rs2[tid] = s2;
  __syncthreads();
  for (int o = 128; o > 0; o >>= 1) {
    if (tid < o) { rs[tid] += rs[tid + o]; rs2[tid] += rs2[tid + o]; }
    __syncthreads();
  }
  const float mean = rs[0] * (1.0f / N_EMBD);
  const float var  = rs2[0] * (1.0f / N_EMBD) - mean * mean;
  const float rstd = rsqrtf(var + 1e-5f);
  for (int i = tid; i < N_EMBD; i += 256)
    out[(size_t)row * N_EMBD + i] = (_Float16)((xr[i] - mean) * rstd * w[i] + b[i]);
}

// ---------------------------------------------------------------------------
// Weight converters (f32 -> f16, transposed so GEMM B reads contiguous K)
// ---------------------------------------------------------------------------
// Wq/Wk/Wv: [H][C][D] f32  ->  BT[n = h*64+d][c] f16  (N=1024, K=1024)
__global__ void cvt_qkv_w_kernel(const float* __restrict__ W, _Float16* __restrict__ BT)
{
  int i = blockIdx.x * 256 + threadIdx.x;            // over 1024*1024
  if (i >= N_EMBD * N_EMBD) return;
  int n = i >> 10, c = i & 1023;
  int h = n >> 6,  d = n & 63;
  BT[i] = (_Float16)W[((size_t)h * N_EMBD + c) * HEAD + d];
}

// generic [R][C] f32 -> [C][R] f16 transpose-convert
__global__ void cvt_transpose_kernel(const float* __restrict__ in, _Float16* __restrict__ out,
                                     int R, int C)
{
  int i = blockIdx.x * 256 + threadIdx.x;
  if (i >= R * C) return;
  int c = i / R, r = i % R;
  out[i] = (_Float16)in[(size_t)r * C + c];
}

// vf32 [(b,t)][h*64+d] -> v16T [bh][d][t] f16 (B-frags of PV read contiguous s)
__global__ void cvt_v_kernel(const float* __restrict__ vf, _Float16* __restrict__ vT)
{
  int i = blockIdx.x * 256 + threadIdx.x;            // over 2*16*64*2048 = 2^22
  if (i >= BATCH * N_EMBD * TSEQ) return;
  int t = i & (TSEQ - 1);
  int d = (i >> 11) & 63;
  int h = (i >> 17) & 15;
  int b = i >> 21;
  vT[i] = (_Float16)vf[((size_t)(b * TSEQ + t)) * N_EMBD + h * HEAD + d];
}

// ---------------------------------------------------------------------------
// RoPE on f32 q/k GEMM outputs + repack to f16 [bh][t][d]
// ---------------------------------------------------------------------------
__global__ void rope_pack_kernel(const float* __restrict__ qf, const float* __restrict__ kf,
                                 _Float16* __restrict__ q16, _Float16* __restrict__ k16)
{
  int tid = blockIdx.x * 256 + threadIdx.x;          // over 2*2048*16*32 = 2^21
  if (tid >= BATCH * TSEQ * N_HEAD * 32) return;
  int i = tid & 31;
  int h = (tid >> 5) & 15;
  int t = (tid >> 9) & (TSEQ - 1);
  int b = tid >> 20;
  int d0 = 2 * i, d1 = 2 * i + 1;
  float f0 = powf(10000.0f, -(float)(2 * (d0 & 31)) * (1.0f / 64.0f));
  float f1 = powf(10000.0f, -(float)(2 * (d1 & 31)) * (1.0f / 64.0f));
  float tf = (float)t;
  float c0 = cosf(tf * f0), s0 = sinf(tf * f0);
  float c1 = cosf(tf * f1), s1 = sinf(tf * f1);
  size_t inBase  = ((size_t)(b * TSEQ + t)) * N_EMBD + h * HEAD;
  size_t outBase = (((size_t)(b * N_HEAD + h)) * TSEQ + t) * HEAD;
  float q0v = qf[inBase + d0], q1v = qf[inBase + d1];
  float k0v = kf[inBase + d0], k1v = kf[inBase + d1];
  q16[outBase + d0] = (_Float16)(q0v * c0 - q1v * s0);
  q16[outBase + d1] = (_Float16)(q1v * c1 + q0v * s1);
  k16[outBase + d0] = (_Float16)(k0v * c0 - k1v * s0);
  k16[outBase + d1] = (_Float16)(k1v * c1 + k0v * s1);
}

// ---------------------------------------------------------------------------
// WMMA GEMM with async global->LDS TRIPLE-buffered B staging.
// out[M][N] f32 = A16[M][K] * BT16[N][K] (+bias)(+resid)
// block = 128 thr (4 waves); block tile 128x64; wave tile 32x64.
// B k-slice (64 n x 32 k halves = 4KB) staged once per block via
// GLOBAL_LOAD_ASYNC_TO_LDS_B128 (ASYNCcnt). Triple buffering => a single
// barrier per k-step (refill target can never alias a buffer a laggard wave
// still reads). All 4 B frags batched -> one s_wait_dscnt, 8 WMMAs burst.
// ---------------------------------------------------------------------------
__global__ void gemm_wmma_kernel(const _Float16* __restrict__ A,
                                 const _Float16* __restrict__ BT,
                                 const float* __restrict__ bias,
                                 const float* __restrict__ resid,
                                 float* __restrict__ out,
                                 int M, int N, int K)
{
  __shared__ __align__(128) _Float16 ldsB[3][64 * 32];   // 3 x 4KB

  const int tid  = threadIdx.x;
  const int lane = tid & 31;
  const int wave = tid >> 5;
  const int hi   = lane >> 4;
  const int ln   = lane & 15;
  const int m0   = blockIdx.y * 128 + wave * 32;
  const int n0   = blockIdx.x * 64;

  const unsigned ldsBase = (unsigned)(uintptr_t)(&ldsB[0][0]);
  const unsigned long long gBase = (unsigned long long)(uintptr_t)BT;

  v8f acc[2][4];
  for (int r2 = 0; r2 < 2; ++r2)
    for (int j = 0; j < 4; ++j)
      for (int r = 0; r < 8; ++r) acc[r2][j][r] = 0.0f;

  // stage one 64x32-half B slice into ldsB[buf] (2 x 16B chunks per thread)
  auto stageB = [&](int buf, int k0) {
#pragma unroll
    for (int c2 = 0; c2 < 2; ++c2) {
      const int c   = tid * 2 + c2;                 // 0..255 chunks of 16B
      const int row = c >> 2;                       // 0..63
      const int sub = c & 3;                        // 4 x 16B per 64B row
      const unsigned ldsOff = ldsBase + (unsigned)(buf * 4096 + row * 64 + sub * 16);
      const unsigned gOff   = ((unsigned)(n0 + row) * (unsigned)K + (unsigned)k0) * 2u
                              + (unsigned)sub * 16u;
      asm volatile("global_load_async_to_lds_b128 %0, %1, %2"
                   :: "v"(ldsOff), "v"(gOff), "s"(gBase)
                   : "memory");
    }
  };

  // A fragment pointers advance by 32 halves per k-step (no per-iter 64b mul)
  const _Float16* aPtr0 = A + (size_t)(m0 + ln) * K + (hi ? 8 : 0);
  const _Float16* aPtr1 = A + (size_t)(m0 + 16 + ln) * K + (hi ? 8 : 0);

  stageB(0, 0);
  int bufCur = 0;
  for (int k0 = 0; k0 < K; k0 += 32) {
    const bool hasNext = (k0 + 32) < K;
    if (hasNext) {
      int bufNxt = bufCur + 1; if (bufNxt == 3) bufNxt = 0;
      stageB(bufNxt, k0 + 32);
    }

    // A fragments direct from global (latency overlaps the async wait)
    FragU a0, a1;
    a0.u[0] = *(const uint4*)(aPtr0);
    a0.u[1] = *(const uint4*)(aPtr0 + 16);
    a1.u[0] = *(const uint4*)(aPtr1);
    a1.u[1] = *(const uint4*)(aPtr1 + 16);
    aPtr0 += 32; aPtr1 += 32;

    if (hasNext) asm volatile("s_wait_asynccnt 0x2" ::: "memory");
    else         asm volatile("s_wait_asynccnt 0x0" ::: "memory");
    __syncthreads();   // single barrier: buffer bufCur fully written everywhere

    // batch all 4 B frags (one DS wait), then 8 back-to-back WMMAs
    FragU b[4];
#pragma unroll
    for (int j = 0; j < 4; ++j) {
      const _Float16* bp = &ldsB[bufCur][(j * 16 + ln) * 32 + (hi ? 8 : 0)];
      b[j].u[0] = *(const uint4*)(bp);
      b[j].u[1] = *(const uint4*)(bp + 16);
    }
#pragma unroll
    for (int j = 0; j < 4; ++j) {
      acc[0][j] = wmma_f16(a0.h, b[j].h, acc[0][j]);
      acc[1][j] = wmma_f16(a1.h, b[j].h, acc[1][j]);
    }

    ++bufCur; if (bufCur == 3) bufCur = 0;
  }

#pragma unroll
  for (int r2 = 0; r2 < 2; ++r2)
#pragma unroll
    for (int j = 0; j < 4; ++j) {
      const int n = n0 + j * 16 + ln;
#pragma unroll
      for (int r = 0; r < 8; ++r) {
        const int m = m0 + r2 * 16 + r + 8 * hi;
        float v = acc[r2][j][r];
        if (bias)  v += bias[n];
        if (resid) v += resid[(size_t)m * N + n];
        out[(size_t)m * N + n] = v;
      }
    }
}

// ---------------------------------------------------------------------------
// Flash attention: one wave per 16-row q-tile of one (b,h). Causal, scale 1/8.
// 32-wide key tiles: QK^T = 4 full WMMAs, PV = 4 full WMMAs (no K-padding).
// Fragment loads batched ahead of each WMMA burst (one wait per burst).
// q16,k16: [bh][t][64] f16 ; vT16: [bh][d][t] f16 ; o16: [(b,t)][h*64+d] f16
// ---------------------------------------------------------------------------
__global__ void attn_kernel(const _Float16* __restrict__ q16,
                            const _Float16* __restrict__ k16,
                            const _Float16* __restrict__ vT16,
                            _Float16* __restrict__ o16)
{
  __shared__ __align__(16) _Float16 pSh[4][16 * 32];   // 1KB per wave
  const int lane = threadIdx.x & 31;
  const int wave = threadIdx.x >> 5;
  const int hi   = lane >> 4;
  const int ln   = lane & 15;
  const int bh   = blockIdx.x;                    // 0..31
  const int qt   = blockIdx.y * 4 + wave;         // 0..127
  const int q0   = qt * 16;

  // Q tile (16x64) -> two A-fragments
  const _Float16* qb = q16 + ((size_t)bh * TSEQ + q0 + ln) * HEAD;
  const int kb = hi ? 8 : 0;
  FragU a0, a1;
  a0.u[0] = *(const uint4*)(qb + kb);       a0.u[1] = *(const uint4*)(qb + kb + 16);
  a1.u[0] = *(const uint4*)(qb + 32 + kb);  a1.u[1] = *(const uint4*)(qb + 32 + kb + 16);

  float Mrow[8], Lrow[8];
  v8f oacc[4];
  for (int j = 0; j < 4; ++j)
    for (int r = 0; r < 8; ++r) oacc[j][r] = 0.0f;
  for (int r = 0; r < 8; ++r) { Mrow[r] = -3.0e38f; Lrow[r] = 0.0f; }

  const float scale = 0.125f;                     // 1/sqrt(64)
  _Float16* ps = pSh[wave];
  const int nst = (qt >> 1) + 1;                  // # of 32-wide key tiles

  for (int st = 0; st < nst; ++st) {
    const int s0 = st * 32;

    // --- batch all K-side fragments, then 4 WMMAs ---
    FragU kb0[2], kb1[2];
#pragma unroll
    for (int g = 0; g < 2; ++g) {
      const _Float16* kbp = k16 + ((size_t)bh * TSEQ + s0 + g * 16 + ln) * HEAD + (hi ? 8 : 0);
      kb0[g].u[0] = *(const uint4*)(kbp);       kb0[g].u[1] = *(const uint4*)(kbp + 16);
      kb1[g].u[0] = *(const uint4*)(kbp + 32);  kb1[g].u[1] = *(const uint4*)(kbp + 48);
    }
    v8f c[2];
    for (int g = 0; g < 2; ++g)
      for (int r = 0; r < 8; ++r) c[g][r] = 0.0f;
#pragma unroll
    for (int g = 0; g < 2; ++g) {
      c[g] = wmma_f16(a0.h, kb0[g].h, c[g]);
      c[g] = wmma_f16(a1.h, kb1[g].h, c[g]);
    }

    float p[2][8];
#pragma unroll
    for (int g = 0; g < 2; ++g) {
      const int scol = s0 + g * 16 + ln;
#pragma unroll
      for (int r = 0; r < 8; ++r) {
        const int tq = q0 + r + 8 * hi;
        float v = c[g][r] * scale;
        if (scol > tq) v = -3.0e38f;              // causal mask
        p[g][r] = v;
      }
    }

    // online softmax over 32 columns
    float alpha[8];
#pragma unroll
    for (int r = 0; r < 8; ++r) {
      float mx = fmaxf(p[0][r], p[1][r]);
      for (int o = 1; o < 16; o <<= 1) mx = fmaxf(mx, __shfl_xor(mx, o, 32));
      const float Mn = fmaxf(Mrow[r], mx);
      const float al = __expf(Mrow[r] - Mn);
      const float pv0 = __expf(p[0][r] - Mn);
      const float pv1 = __expf(p[1][r] - Mn);
      float sum = pv0 + pv1;
      for (int o = 1; o < 16; o <<= 1) sum += __shfl_xor(sum, o, 32);
      Lrow[r] = Lrow[r] * al + sum;
      Mrow[r] = Mn;
      alpha[r] = al;
      p[0][r] = pv0;
      p[1][r] = pv1;
    }
#pragma unroll
    for (int j = 0; j < 4; ++j)
#pragma unroll
      for (int r = 0; r < 8; ++r) oacc[j][r] *= alpha[r];

    // transpose P (16x32, C-layout -> A-layout) through LDS
#pragma unroll
    for (int g = 0; g < 2; ++g)
#pragma unroll
      for (int r = 0; r < 8; ++r)
        ps[(r + 8 * hi) * 32 + g * 16 + ln] = (_Float16)p[g][r];
    asm volatile("s_wait_dscnt 0x0" ::: "memory");
    FragU pa;
    const _Float16* pp = ps + ln * 32 + (hi ? 8 : 0);
    pa.u[0] = *(const uint4*)(pp);        // k = 0..7  (s0.. / s8..)
    pa.u[1] = *(const uint4*)(pp + 16);   // k = 16..23 (s16.. / s24..)

    // --- batch all 4 V fragments, then 4 back-to-back WMMAs ---
    FragU bv[4];
#pragma unroll
    for (int j = 0; j < 4; ++j) {
      const _Float16* vp = vT16 + ((size_t)bh * HEAD + j * 16 + ln) * TSEQ + s0 + (hi ? 8 : 0);
      bv[j].u[0] = *(const uint4*)(vp);
      bv[j].u[1] = *(const uint4*)(vp + 16);
    }
#pragma unroll
    for (int j = 0; j < 4; ++j)
      oacc[j] = wmma_f16(pa.h, bv[j].h, oacc[j]);
  }

  // write O (f16) to [(b,t)][h*64+d] for the projection GEMM
  const int b = bh >> 4, h = bh & 15;
#pragma unroll
  for (int j = 0; j < 4; ++j) {
    const int col = h * HEAD + j * 16 + ln;
#pragma unroll
    for (int r = 0; r < 8; ++r) {
      const int t = q0 + r + 8 * hi;
      o16[((size_t)(b * TSEQ + t)) * N_EMBD + col] = (_Float16)(oacc[j][r] / Lrow[r]);
    }
  }
}

// ---------------------------------------------------------------------------
// SwiGLU-ish gate: g = (a * sigmoid(a)) * b  -> f16
// ---------------------------------------------------------------------------
__global__ void silu_mul_kernel(const float* __restrict__ a, const float* __restrict__ b,
                                _Float16* __restrict__ g)
{
  int i = blockIdx.x * 256 + threadIdx.x;          // over 4096*2048 = 2^23
  if (i >= ROWS * HIDDEN) return;
  float av = a[i];
  float s = av / (1.0f + __expf(-av));
  g[i] = (_Float16)(s * b[i]);
}

// ---------------------------------------------------------------------------
extern "C" void kernel_launch(void* const* d_in, const int* in_sizes, int n_in,
                              void* d_out, int out_size, void* d_ws, size_t ws_size,
                              hipStream_t stream)
{
  (void)in_sizes; (void)n_in; (void)out_size; (void)ws_size;
  const float* x     = (const float*)d_in[0];
  const float* ln1w  = (const float*)d_in[1];
  const float* ln1b  = (const float*)d_in[2];
  const float* ln2w  = (const float*)d_in[3];
  const float* ln2b  = (const float*)d_in[4];
  const float* Wq    = (const float*)d_in[5];
  const float* Wk    = (const float*)d_in[6];
  const float* Wv    = (const float*)d_in[7];
  const float* projw = (const float*)d_in[8];
  const float* projb = (const float*)d_in[9];
  const float* ww    = (const float*)d_in[10];
  const float* wb    = (const float*)d_in[11];
  const float* vw    = (const float*)d_in[12];
  const float* vb    = (const float*)d_in[13];
  const float* pw    = (const float*)d_in[14];
  const float* pb    = (const float*)d_in[15];
  float* outp = (float*)d_out;   // also used as x2 residual buffer (in-place safe)

  char* ws = (char*)d_ws;
  const size_t MB = 1024ull * 1024ull;
  // persistent f16 weights (transposed): [0, 20MB)
  _Float16* WqT   = (_Float16*)(ws + 0 * MB);    // 2MB
  _Float16* WkT   = (_Float16*)(ws + 2 * MB);    // 2MB
  _Float16* WvT   = (_Float16*)(ws + 4 * MB);    // 2MB
  _Float16* projT = (_Float16*)(ws + 6 * MB);    // 2MB
  _Float16* wwT   = (_Float16*)(ws + 8 * MB);    // 4MB
  _Float16* vwT   = (_Float16*)(ws + 12 * MB);   // 4MB
  _Float16* pwT   = (_Float16*)(ws + 16 * MB);   // 4MB
  // activations (lifetime-overlapped)
  _Float16* h16   = (_Float16*)(ws + 20 * MB);   // 8MB  (dead after QKV GEMMs)
  float*    qf    = (float*)   (ws + 28 * MB);   // 16MB (dead after rope)
  float*    kf    = (float*)   (ws + 44 * MB);   // 16MB (dead after rope)
  float*    vf    = (float*)   (ws + 60 * MB);   // 16MB (dead after cvt_v)
  _Float16* q16   = (_Float16*)(ws + 76 * MB);   // 8MB  (dead after attn)
  _Float16* k16   = (_Float16*)(ws + 84 * MB);   // 8MB  (dead after attn)
  _Float16* vT16  = (_Float16*)(ws + 92 * MB);   // 8MB  (dead after attn)
  _Float16* o16   = (_Float16*)(ws + 20 * MB);   // reuse h16 slot
  float*    aF    = (float*)   (ws + 28 * MB);   // 32MB, reuse qf+kf
  float*    bF    = (float*)   (ws + 60 * MB);   // 32MB, reuse vf+q16+k16
  _Float16* h2_16 = (_Float16*)(ws + 100 * MB);  // 8MB
  _Float16* g16   = (_Float16*)(ws + 108 * MB);  // 16MB   (total 124MB)

  // ---- weight conversion ----
  cvt_qkv_w_kernel<<<4096, 256, 0, stream>>>(Wq, WqT);
  cvt_qkv_w_kernel<<<4096, 256, 0, stream>>>(Wk, WkT);
  cvt_qkv_w_kernel<<<4096, 256, 0, stream>>>(Wv, WvT);
  cvt_transpose_kernel<<<4096, 256, 0, stream>>>(projw, projT, N_EMBD, N_EMBD);
  cvt_transpose_kernel<<<8192, 256, 0, stream>>>(ww, wwT, N_EMBD, HIDDEN);
  cvt_transpose_kernel<<<8192, 256, 0, stream>>>(vw, vwT, N_EMBD, HIDDEN);
  cvt_transpose_kernel<<<8192, 256, 0, stream>>>(pw, pwT, HIDDEN, N_EMBD);

  // ---- LN1 ----
  ln_kernel<<<ROWS, 256, 0, stream>>>(x, ln1w, ln1b, h16);

  // ---- QKV GEMMs (M=4096, N=1024, K=1024); block tile 128x64 ----
  dim3 gB(128);
  gemm_wmma_kernel<<<dim3(16, 32), gB, 0, stream>>>(h16, WqT, nullptr, nullptr, qf, ROWS, N_EMBD, N_EMBD);
  gemm_wmma_kernel<<<dim3(16, 32), gB, 0, stream>>>(h16, WkT, nullptr, nullptr, kf, ROWS, N_EMBD, N_EMBD);
  gemm_wmma_kernel<<<dim3(16, 32), gB, 0, stream>>>(h16, WvT, nullptr, nullptr, vf, ROWS, N_EMBD, N_EMBD);

  // ---- RoPE + pack f16, V transpose-pack ----
  rope_pack_kernel<<<8192, 256, 0, stream>>>(qf, kf, q16, k16);
  cvt_v_kernel<<<16384, 256, 0, stream>>>(vf, vT16);

  // ---- flash attention (32 bh x 128 q-tiles, 4 waves/block) ----
  attn_kernel<<<dim3(32, 32), dim3(128), 0, stream>>>(q16, k16, vT16, o16);

  // ---- output projection + residual: x2 = x + o@proj_w + proj_b  (into d_out) ----
  gemm_wmma_kernel<<<dim3(16, 32), gB, 0, stream>>>(o16, projT, projb, x, outp, ROWS, N_EMBD, N_EMBD);

  // ---- LN2 ----
  ln_kernel<<<ROWS, 256, 0, stream>>>(outp, ln2w, ln2b, h2_16);

  // ---- MLP: a = h2@w_w + w_b ; b = h2@v_w + v_b ----
  gemm_wmma_kernel<<<dim3(32, 32), gB, 0, stream>>>(h2_16, wwT, wb, nullptr, aF, ROWS, HIDDEN, N_EMBD);
  gemm_wmma_kernel<<<dim3(32, 32), gB, 0, stream>>>(h2_16, vwT, vb, nullptr, bF, ROWS, HIDDEN, N_EMBD);

  // ---- gate: g = silu(a)*b ----
  silu_mul_kernel<<<32768, 256, 0, stream>>>(aF, bF, g16);

  // ---- out = x2 + g@p_w + p_b  (in-place read-then-write on d_out) ----
  gemm_wmma_kernel<<<dim3(16, 32), gB, 0, stream>>>(g16, pwT, pb, outp, outp, ROWS, N_EMBD, HIDDEN);
}
